// RNN_24515673325781
// MI455X (gfx1250) — matrix operand order repaired and checked by reading, not compiled, not run on hardware
//
#include <hip/hip_runtime.h>
#include <hip/hip_bf16.h>

#define N_INPUT   64
#define N_HIDDEN  512
#define N_OUTPUT  24
#define BATCH     256
#define TSTEPS    512
#define BTILE     32          // 2 WMMA row tiles per block -> B fragments reused 2x

#if __has_builtin(__builtin_amdgcn_tanhf)
#define FAST_TANH(v) __builtin_amdgcn_tanhf(v)
#else
#define FAST_TANH(v) tanhf(v)
#endif

typedef __attribute__((ext_vector_type(16))) __bf16 v16bf;
typedef __attribute__((ext_vector_type(8)))  float  v8f;

union FragU { v16bf v; uint4 u[2]; };

// Dense 16-bit B-matrix (K x N = 32x16) fragment from row-major W[row][k]:
// lanes 0-15 hold column N=lane, K = kc*32 + 0..15 ; lanes 16-31 hold K = kc*32 + 16..31.
__device__ __forceinline__ v16bf load_b_frag(const __bf16* p /* row base + hi*16 */,
                                             int kc) {
  FragU f;
  f.u[0] = *(const uint4*)(p + kc * 32);
  f.u[1] = *(const uint4*)(p + kc * 32 + 8);
  return f.v;
}

// 16-bit A-matrix (M x K = 16x32) fragment, contiguous k order (same as B):
// lanes 0-15: M=lane, K = kc*32 + 0..15; lanes 16-31: M=lane-16, K = kc*32 + 16..31
__device__ __forceinline__ v16bf load_a_frag(const __bf16* p /* row base + hi*16 */, int kc) {
  FragU f;
  f.u[0] = *(const uint4*)(p + kc * 32);
  f.u[1] = *(const uint4*)(p + kc * 32 + 8);
  return f.v;
}

// ---- prologue: convert weights f32 -> bf16 into workspace (W_ho zero-padded to 32 rows) ----
__global__ void cvt_weights(const float* __restrict__ Wih, const float* __restrict__ Whh,
                            const float* __restrict__ Who,
                            __bf16* __restrict__ wih_b, __bf16* __restrict__ whh_b,
                            __bf16* __restrict__ who_b) {
  int i0 = blockIdx.x * blockDim.x + threadIdx.x;
  int stride = gridDim.x * blockDim.x;
  for (int k = i0; k < N_HIDDEN * N_HIDDEN; k += stride) whh_b[k] = (__bf16)Whh[k];
  for (int k = i0; k < N_HIDDEN * N_INPUT;  k += stride) wih_b[k] = (__bf16)Wih[k];
  for (int k = i0; k < 32 * N_HIDDEN; k += stride)
    who_b[k] = (k < N_OUTPUT * N_HIDDEN) ? (__bf16)Who[k] : (__bf16)0.0f;
}

// ---- persistent recurrent kernel: one block owns 32 batch rows for all 512 steps ----
__global__ __launch_bounds__(1024)
void rnn_persistent(const float* __restrict__ x,
                    const __bf16* __restrict__ whh,
                    const __bf16* __restrict__ wih,
                    const __bf16* __restrict__ who,
                    const float* __restrict__ b_ih,
                    const float* __restrict__ b_hh,
                    const float* __restrict__ b_ho,
                    float* __restrict__ out) {
  __shared__ __align__(16) __bf16 h_lds[BTILE * N_HIDDEN];   // 32 KB hidden state (bf16)
  __shared__ __align__(16) __bf16 xbuf[BTILE * N_INPUT];     // 4 KB x_t staging (bf16)
  __shared__ float obuf[2 * BTILE * 16];                     // 4 KB split-K reduction

  const int tid    = threadIdx.x;
  const int w      = tid >> 5;      // wave id 0..31 -> owns hidden column tile w (16 cols)
  const int lane   = tid & 31;
  const int laneLo = lane & 15;
  const int hi     = lane >> 4;
  const int bb     = blockIdx.x * BTILE;

  // Per-lane base pointer for streaming W_hh B-fragments from L2 each step.
  const __bf16* pB_hh = whh + (size_t)(w * 16 + laneLo) * N_HIDDEN + hi * 16;

  // Small register-resident weights: W_ih (2 frags) + W_ho (1 frag) = 24 VGPRs.
  v16bf wih_f[2];
  #pragma unroll
  for (int kc = 0; kc < 2; ++kc)
    wih_f[kc] = load_b_frag(wih + (size_t)(w * 16 + laneLo) * N_INPUT + hi * 16, kc);

  const int otile = w >> 4;   // output-head tile this wave feeds (0 or 1)
  const int okc   = w & 15;   // split-K chunk (32 of 512 k-values)
  const v16bf who_f =
      load_b_frag(who + (size_t)(otile * 16 + laneLo) * N_HIDDEN + hi * 16, okc);

  const float bias_v = b_ih[w * 16 + laneLo] + b_hh[w * 16 + laneLo];

  // store duty (waves 0..3): wave -> (tile = w&1, row half = w>>1)
  const int   st_n   = (w < 4) ? ((w & 1) * 16 + laneLo) : 64;
  const int   st_mb  = (w >> 1) * 16;
  const float st_bho = (st_n < N_OUTPUT) ? b_ho[st_n] : 0.0f;

  // A-fragment LDS base pointers for the two row tiles
  const __bf16* pA0 = h_lds + laneLo * N_HIDDEN + hi * 16;
  const __bf16* pA1 = pA0 + 16 * N_HIDDEN;
  const __bf16* pX0 = xbuf + laneLo * N_INPUT + hi * 16;
  const __bf16* pX1 = pX0 + 16 * N_INPUT;

  for (int i = tid; i < BTILE * N_HIDDEN; i += blockDim.x) h_lds[i] = (__bf16)0.0f;
  __syncthreads();

  for (int t = 0; t < TSTEPS; ++t) {
    // Opaque *offset* (not pointer): the empty volatile asm is re-executed every
    // timestep so LICM cannot hoist the 16 B-fragment loads out of the t-loop
    // (that previously pinned 128 VGPRs and spilled to scratch). Keeping the
    // base pointer untouched preserves its global address space, so these
    // lower to global_load_b128 (LOADcnt only) instead of flat_load_b128
    // (which would also tie up DScnt and couple with the LDS A-path waits).
    int b_off = 0;
    asm volatile("" : "+v"(b_off));
    const __bf16* pB = pB_hh + b_off;

    // stage x[:, t, :] tile (32x64 f32 -> bf16); zero reduction buffer;
    // prefetch next timestep's x rows (global_prefetch_b8)
    for (int e = tid; e < BTILE * N_INPUT; e += blockDim.x) {
      int m = e >> 6, i = e & 63;
      size_t xi = (((size_t)(bb + m)) * TSTEPS + t) * N_INPUT + i;
      xbuf[e] = (__bf16)x[xi];
      if (((i & 15) == 0) && (t + 1 < TSTEPS))
        __builtin_prefetch(&x[xi + N_INPUT], 0, 1);
    }
    obuf[tid] = 0.0f;   // 1024 threads zero 2*32*16 floats
    __syncthreads();

    // acc = bias + x_t @ W_ih^T + h @ W_hh^T  (two 16-row tiles, one 16-col tile)
    v8f acc0 = { bias_v, bias_v, bias_v, bias_v, bias_v, bias_v, bias_v, bias_v };
    v8f acc1 = acc0;
    #pragma unroll
    for (int kc = 0; kc < 2; ++kc) {
      v16bf a0 = load_a_frag(pX0, kc);
      v16bf a1 = load_a_frag(pX1, kc);
      acc0 = __builtin_amdgcn_wmma_f32_16x16x32_bf16(false, a0, false, wih_f[kc],
                                                     (short)0, acc0, false, false);
      acc1 = __builtin_amdgcn_wmma_f32_16x16x32_bf16(false, a1, false, wih_f[kc],
                                                     (short)0, acc1, false, false);
    }
    #pragma unroll
    for (int kc = 0; kc < 16; ++kc) {
      v16bf b  = load_b_frag(pB, kc);        // global_load_b128 x2 (L2 hit)
      v16bf a0 = load_a_frag(pA0, kc);
      v16bf a1 = load_a_frag(pA1, kc);
      acc0 = __builtin_amdgcn_wmma_f32_16x16x32_bf16(false, a0, false, b,
                                                     (short)0, acc0, false, false);
      acc1 = __builtin_amdgcn_wmma_f32_16x16x32_bf16(false, a1, false, b,
                                                     (short)0, acc1, false, false);
    }
    __syncthreads();   // all reads of old h done

    // h = tanh(acc) -> LDS   (C/D layout: VGPR r -> M = r + hi*8, N = laneLo)
    #pragma unroll
    for (int r = 0; r < 8; ++r) {
      int m = r + hi * 8;
      h_lds[m * N_HIDDEN + w * 16 + laneLo]        = (__bf16)FAST_TANH(acc0[r]);
      h_lds[(m + 16) * N_HIDDEN + w * 16 + laneLo] = (__bf16)FAST_TANH(acc1[r]);
    }
    __syncthreads();   // new h visible

    // output head: split-K WMMA (16 waves per output tile), ds_add_f32 reduction
    {
      v16bf a0 = load_a_frag(pA0, okc);
      v16bf a1 = load_a_frag(pA1, okc);
      v8f z = { 0.f, 0.f, 0.f, 0.f, 0.f, 0.f, 0.f, 0.f };
      v8f o0 = __builtin_amdgcn_wmma_f32_16x16x32_bf16(false, a0, false, who_f,
                                                       (short)0, z, false, false);
      v8f o1 = __builtin_amdgcn_wmma_f32_16x16x32_bf16(false, a1, false, who_f,
                                                       (short)0, z, false, false);
      #pragma unroll
      for (int r = 0; r < 8; ++r) {
        int m = r + hi * 8;
        atomicAdd(&obuf[otile * (BTILE * 16) + m * 16 + laneLo], o0[r]);
        atomicAdd(&obuf[otile * (BTILE * 16) + (m + 16) * 16 + laneLo], o1[r]);
      }
    }
    __syncthreads();

    // waves 0..3: apply bias + tanh, scatter-store to out[B,T,24]
    if (st_n < N_OUTPUT) {
      #pragma unroll
      for (int r = 0; r < 8; ++r) {
        int m = st_mb + r + hi * 8;
        float v = FAST_TANH(obuf[(w & 1) * (BTILE * 16) + m * 16 + laneLo] + st_bho);
        out[(((size_t)(bb + m)) * TSTEPS + t) * N_OUTPUT + st_n] = v;
      }
    }
    __syncthreads();
  }
}

extern "C" void kernel_launch(void* const* d_in, const int* in_sizes, int n_in,
                              void* d_out, int out_size, void* d_ws, size_t ws_size,
                              hipStream_t stream) {
  const float* x   = (const float*)d_in[0];
  const float* Wih = (const float*)d_in[1];
  const float* Whh = (const float*)d_in[2];
  const float* bih = (const float*)d_in[3];
  const float* bhh = (const float*)d_in[4];
  const float* Who = (const float*)d_in[5];
  const float* bho = (const float*)d_in[6];
  float* out = (float*)d_out;

  __bf16* whh_b = (__bf16*)d_ws;                       // 512*512*2 = 512 KB
  __bf16* wih_b = whh_b + N_HIDDEN * N_HIDDEN;         // 512*64*2  =  64 KB
  __bf16* who_b = wih_b + N_HIDDEN * N_INPUT;          // 32*512*2  =  32 KB (padded)

  cvt_weights<<<128, 256, 0, stream>>>(Wih, Whh, Who, wih_b, whh_b, who_b);
  rnn_persistent<<<BATCH / BTILE, 1024, 0, stream>>>(x, whh_b, wih_b, who_b,
                                                     bih, bhh, bho, out);
}
